// MSGDC_15719580304514
// MI455X (gfx1250) — 8-bit matrix operand order repaired and checked by reading, compiled, not run on hardware
//
#include <hip/hip_runtime.h>
#include <hip/hip_bf16.h>

typedef __attribute__((ext_vector_type(8))) int   v8i;
typedef __attribute__((ext_vector_type(4))) int   v4i;

union AFrag { v8i v; unsigned long long q[4]; };
union BFrag { v8i v; v4i h[2]; };

#define BATCH 16
#define CHAN  256
#define HH    56
#define WW    56
#define HWSZ  (HH*WW)
#define GRP   4
#define CG    64
#define HP    66
#define WP    66
#define NBR   3

#if defined(__gfx1250__) && \
    __has_builtin(__builtin_amdgcn_global_load_async_to_lds_b128) && \
    __has_builtin(__builtin_amdgcn_s_wait_asynccnt)
#define USE_ASYNC_LDS 1
typedef __attribute__((address_space(1))) v4i gas_v4i;
typedef __attribute__((address_space(3))) v4i las_v4i;
#else
#define USE_ASYNC_LDS 0
#endif

__global__ void binq_kernel(const float* __restrict__ x,
                            const float* __restrict__ bias,
                            signed char* __restrict__ xq) {
    size_t i = (size_t)blockIdx.x * 256 + threadIdx.x;
    if (i >= (size_t)BATCH * CHAN * HWSZ) return;
    int w = (int)(i % WW);
    int h = (int)((i / WW) % HH);
    int c = (int)((i / HWSZ) % CHAN);
    int b = (int)(i / ((size_t)HWSZ * CHAN));
    float v = x[i] + bias[c];
    signed char s = (v >= 0.f) ? (signed char)1 : (signed char)-1;
    size_t dst = ((((size_t)(b * GRP + (c >> 6)) * HP + (h + 5)) * WP) + (w + 5)) * CG + (c & 63);
    xq[dst] = s;
}

__global__ void binw_kernel(const float* __restrict__ w,
                            signed char* __restrict__ wpk,
                            float* __restrict__ scale) {
    const int br = blockIdx.x;
    const int o  = threadIdx.x;
    const float* ws = w + ((size_t)(br * CHAN + o)) * (CG * 9);
    float s = 0.f, sa = 0.f;
    for (int k = 0; k < CG * 9; ++k) {
        float v = ws[k];
        s  += v;
        sa += fabsf(v);
    }
    const float mean = s * (1.f / (CG * 9));
    scale[br * CHAN + o] = sa * (1.f / (CG * 9));
    const int g = o >> 6, n = o & 63;
    for (int ci = 0; ci < CG; ++ci) {
        for (int kk = 0; kk < 9; ++kk) {
            float v = ws[ci * 9 + kk] - mean;
            size_t dst = (((size_t)((br * GRP + g) * 9 + kk) * CG) + n) * CG + ci;
            wpk[dst] = (v >= 0.f) ? (signed char)1 : (signed char)-1;
        }
    }
}

__global__ __launch_bounds__(128) void conv_kernel(
        const signed char* __restrict__ xq,
        const signed char* __restrict__ wpk,
        const float* __restrict__ scale,
        const float* __restrict__ cb,
        const float* __restrict__ move1,
        const float* __restrict__ alpha,
        const float* __restrict__ move2,
        float* __restrict__ outpre) {
    __shared__ __align__(16) signed char Bsm[9 * CG * CG];

    const int tid    = threadIdx.x;
    const int lane   = tid & 31;
    const int wv     = tid >> 5;
    const int p0     = blockIdx.x * 64;
    const int g      = blockIdx.y;
    const int b      = blockIdx.z;
    const int row    = lane & 15;
    const int half   = lane >> 4;
    const int colSel = lane & 15;

    const int p = p0 + wv * 16 + row;
    const int h = p / WW, w = p % WW;
    const signed char* abase = xq
        + (((size_t)(b * GRP + g) * HP + (h + 5)) * WP + (w + 5)) * CG + half * 16;

    float outf[4][8];
#pragma unroll
    for (int nt = 0; nt < 4; ++nt)
#pragma unroll
        for (int e = 0; e < 8; ++e) outf[nt][e] = 0.f;

    const int DIL[NBR] = {1, 3, 5};

#pragma unroll
    for (int br = 0; br < NBR; ++br) {
        const v4i* wsrc = (const v4i*)(wpk + ((size_t)(br * GRP + g) * 9) * (CG * CG));
        __syncthreads();
#if USE_ASYNC_LDS
        for (int it = tid; it < (9 * CG * CG) / 16; it += 128) {
            __builtin_amdgcn_global_load_async_to_lds_b128(
                (gas_v4i*)(wsrc + it),
                (las_v4i*)((v4i*)Bsm + it),
                0, 0);
        }
        __builtin_amdgcn_s_wait_asynccnt(0);
#else
        for (int it = tid; it < (9 * CG * CG) / 16; it += 128)
            ((v4i*)Bsm)[it] = wsrc[it];
#endif
        if (br + 1 < NBR)
            __builtin_prefetch(wpk + ((size_t)((br + 1) * GRP + g) * 9) * (CG * CG), 0, 1);
        __syncthreads();

        const int d = DIL[br];
        v8i acc[4];
#pragma unroll
        for (int nt = 0; nt < 4; ++nt) acc[nt] = (v8i){0, 0, 0, 0, 0, 0, 0, 0};

#pragma unroll
        for (int ky = 0; ky < 3; ++ky) {
#pragma unroll
            for (int kx = 0; kx < 3; ++kx) {
                const int tapoff = (((ky - 1) * d) * WP + (kx - 1) * d) * CG;
                AFrag a;
                const unsigned long long* ap =
                    (const unsigned long long*)(abase + tapoff);
                a.q[0] = ap[0];
                a.q[1] = ap[1];
                a.q[2] = ap[4];
                a.q[3] = ap[5];
                const int tap = ky * 3 + kx;
#pragma unroll
                for (int nt = 0; nt < 4; ++nt) {
                    BFrag bf;
                    const v4i* bp = (const v4i*)(Bsm + tap * (CG * CG)
                                                 + (nt * 16 + colSel) * CG + half * 16);
                    bf.h[0] = bp[0];
                    bf.h[1] = bp[2];
                    acc[nt] = __builtin_amdgcn_wmma_i32_16x16x64_iu8(
                        true, a.v, true, bf.v, acc[nt], false, false);
                }
            }
        }

#pragma unroll
        for (int nt = 0; nt < 4; ++nt) {
            const int o  = g * CG + nt * 16 + colSel;
            const float sc = scale[br * CHAN + o];
            const float cc = cb[br * CHAN + o];
            const float m1 = move1[br * CHAN + o];
            const float al = alpha[br * CHAN + o];
            const float m2 = move2[br * CHAN + o];
#pragma unroll
            for (int e = 0; e < 8; ++e) {
                float y = sc * (float)acc[nt][e] + cc - m1;
                y = (y >= 0.f) ? y : al * y;
                outf[nt][e] += y + m2;
            }
        }
    }

#pragma unroll
    for (int nt = 0; nt < 4; ++nt) {
        const int o = g * CG + nt * 16 + colSel;
#pragma unroll
        for (int e = 0; e < 8; ++e) {
            const int pe = p0 + wv * 16 + half * 8 + e;
            outpre[((size_t)b * HWSZ + pe) * CHAN + o] = outf[nt][e];
        }
    }
}

__global__ __launch_bounds__(256) void ln_kernel(const float* __restrict__ pre,
                                                 const float* __restrict__ gamma,
                                                 const float* __restrict__ beta,
                                                 float* __restrict__ out) {
    __shared__ float sm[16][CHAN + 1];
    __shared__ float red0[16][16];
    __shared__ float red1[16][16];
    __shared__ float mean_s[16];
    __shared__ float rstd_s[16];

    const int tile = blockIdx.x;
    const int b    = tile / (HWSZ / 16);
    const int p0   = (tile % (HWSZ / 16)) * 16;
    const int tid  = threadIdx.x;

    for (int t = 0; t < 16; ++t)
        sm[t][tid] = pre[((size_t)b * HWSZ + p0 + t) * CHAN + tid];
    __syncthreads();

    const int pp = tid >> 4, sub = tid & 15;
    float s = 0.f, s2 = 0.f;
#pragma unroll
    for (int k = 0; k < 16; ++k) {
        float v = sm[pp][sub + k * 16];
        s  += v;
        s2 += v * v;
    }
    red0[pp][sub] = s;
    red1[pp][sub] = s2;
    __syncthreads();
    if (sub == 0) {
        float a = 0.f, q = 0.f;
#pragma unroll
        for (int k = 0; k < 16; ++k) { a += red0[pp][k]; q += red1[pp][k]; }
        const float m   = a * (1.f / CHAN);
        const float var = q * (1.f / CHAN) - m * m;
        mean_s[pp] = m;
        rstd_s[pp] = rsqrtf(var + 1e-5f);
    }
    __syncthreads();

#pragma unroll
    for (int k = 0; k < 16; ++k) {
        const int c = (tid >> 4) + k * 16;
        const int t = tid & 15;
        const float v = (sm[t][c] - mean_s[t]) * rstd_s[t] * gamma[c] + beta[c];
        out[((size_t)(b * CHAN + c)) * HWSZ + p0 + t] = v;
    }
}

extern "C" void kernel_launch(void* const* d_in, const int* in_sizes, int n_in,
                              void* d_out, int out_size, void* d_ws, size_t ws_size,
                              hipStream_t stream) {
    const float* x     = (const float*)d_in[0];
    const float* bias  = (const float*)d_in[1];
    const float* w     = (const float*)d_in[2];
    const float* cb    = (const float*)d_in[3];
    const float* move1 = (const float*)d_in[4];
    const float* alpha = (const float*)d_in[5];
    const float* move2 = (const float*)d_in[6];
    const float* gamma = (const float*)d_in[7];
    const float* beta  = (const float*)d_in[8];

    char* ws = (char*)d_ws;
    const size_t XQ_BYTES = (size_t)BATCH * GRP * HP * WP * CG;
    const size_t WP_BYTES = (size_t)NBR * GRP * 9 * CG * CG;
    const size_t SC_BYTES = 4096;
    signed char* xq   = (signed char*)ws;
    signed char* wpk  = (signed char*)(ws + XQ_BYTES);
    float*       scl  = (float*)(ws + XQ_BYTES + WP_BYTES);
    float*       pre  = (float*)(ws + XQ_BYTES + WP_BYTES + SC_BYTES);

    (void)hipMemsetAsync(xq, 0, XQ_BYTES, stream);

    binq_kernel<<<(BATCH * CHAN * HWSZ) / 256, 256, 0, stream>>>(x, bias, xq);
    binw_kernel<<<NBR, CHAN, 0, stream>>>(w, wpk, scl);
    conv_kernel<<<dim3(HWSZ / 64, GRP, BATCH), 128, 0, stream>>>(
        xq, wpk, scl, cb, move1, alpha, move2, pre);
    ln_kernel<<<BATCH * (HWSZ / 16), 256, 0, stream>>>(pre, gamma, beta, (float*)d_out);
}
